// NFM_50663434224284
// MI455X (gfx1250) — matrix operand order repaired and checked by reading, not compiled, not run on hardware
//
#include <hip/hip_runtime.h>
#include <hip/hip_bf16.h>
#include <math.h>

// ---------------- problem sizes ----------------
#define B_SZ  16384
#define DS    13
#define SP    26
#define NCAT  100000
#define EDIM  32
#define H1N   256
#define H2N   128
#define KP1   64          // padded K for GEMM1 (45 -> 64)

// ---------------- WMMA types (CDNA5 wave32) ----------------
typedef _Float16 h8   __attribute__((ext_vector_type(8)));    // 16 bytes
typedef _Float16 v16h __attribute__((ext_vector_type(16)));   // 8 VGPRs
typedef float    v8f  __attribute__((ext_vector_type(8)));    // 8 VGPRs

union ABFrag { v16h v; h8 h[2]; };

__device__ __forceinline__ v8f wmma_f16(const ABFrag& a, const ABFrag& b, v8f c) {
  // D = A(16x32 f16) x B(32x16 f16) + C(16x16 f32)
  return __builtin_amdgcn_wmma_f32_16x16x32_f16(
      /*neg_a=*/false, a.v, /*neg_b=*/false, b.v,
      /*c_mod=*/(short)0, c, /*reuse_a=*/false, /*reuse_b=*/false);
}

// ---------------- weight conversion ----------------
__global__ __launch_bounds__(256) void nfm_cvt_w1(const float* __restrict__ w1,
                                                  _Float16* __restrict__ w1h) {
  int i = blockIdx.x * blockDim.x + threadIdx.x;        // 256*64
  int n = i >> 6, k = i & 63;
  w1h[i] = (_Float16)((k < EDIM + DS) ? w1[n * (EDIM + DS) + k] : 0.0f);
}

__global__ __launch_bounds__(256) void nfm_cvt_w2(const float* __restrict__ w2,
                                                  _Float16* __restrict__ w2h) {
  int i = blockIdx.x * blockDim.x + threadIdx.x;        // 128*256
  w2h[i] = (_Float16)w2[i];
}

// ---------------- phase 1: gathers + bi-interaction + linear part ----------------
// one wave32 per batch row; lane = embedding dim (0..31) -> each field gather is
// a fully coalesced 128B line straight from HBM. This phase is the roofline
// bottleneck (~54 MB of gathers).
__global__ __launch_bounds__(256) void nfm_features(
    const float* __restrict__ ds,  const int*   __restrict__ sp,
    const float* __restrict__ emb, const float* __restrict__ lint,
    const float* __restrict__ linw, const float* __restrict__ linb,
    _Float16* __restrict__ X, float* __restrict__ lin_out) {
  const int lane = threadIdx.x & 31;
  const int row  = (blockIdx.x * blockDim.x + threadIdx.x) >> 5;
  const int* sprow = sp + row * SP;

  float s = 0.0f, sq = 0.0f;
  #pragma unroll 2
  for (int f = 0; f < SP; ++f) {
    const int idx = sprow[f];
    if (f + 1 < SP) {  // hide HBM latency of the next gather (global_prefetch_b8)
      __builtin_prefetch(emb + ((size_t)(f + 1) * NCAT + (size_t)sprow[f + 1]) * EDIM, 0, 0);
    }
    const float e = emb[((size_t)f * NCAT + (size_t)idx) * EDIM + lane];
    s  += e;
    sq += e * e;
  }
  const float bi = 0.5f * (s * s - sq);

  // x row: [ bi(32) | ds(13) | zeros(19) ], f16, stride KP1
  _Float16* xr = X + (size_t)row * KP1;
  xr[lane] = (_Float16)bi;

  float dsv = 0.0f;
  if (lane < DS) dsv = ds[(size_t)row * DS + lane];
  xr[EDIM + lane] = (_Float16)((lane < DS) ? dsv : 0.0f);

  // linear part: sum of 26 per-field scalars + ds . lin_w + lin_b
  float acc = 0.0f;
  if (lane < SP) acc  = lint[(size_t)lane * NCAT + (size_t)sprow[lane]];
  if (lane < DS) acc += dsv * linw[lane];
  #pragma unroll
  for (int o = 16; o > 0; o >>= 1) acc += __shfl_xor(acc, o, 32);
  if (lane == 0) lin_out[row] = acc + linb[0];
}

// ---------------- phase 2: fully fused MLP ----------------
// One block (8 waves) per 16-row batch tile. H1 (16x256) lives only in LDS;
// H2 never leaves registers; the final 128-dim dot + sigmoid is reduced via
// half-wave shuffles + ds_add_f32. No MLP intermediates touch HBM.
__global__ __launch_bounds__(256) void nfm_mlp(
    const _Float16* __restrict__ X,  const _Float16* __restrict__ W1,
    const float* __restrict__ b1,    const _Float16* __restrict__ W2,
    const float* __restrict__ b2,    const float* __restrict__ fw,
    const float* __restrict__ fb,    const float* __restrict__ lin_out,
    float* __restrict__ out) {
  constexpr int LD1 = H1N + 16;                 // padded stride (halves) vs bank conflicts
  __shared__ __align__(16) _Float16 H1s[16 * LD1];
  __shared__ float acc[16];

  const int lane = threadIdx.x & 31;
  const int w    = threadIdx.x >> 5;            // wave 0..7
  const int m0   = blockIdx.x * 16;
  const int hi   = lane >> 4, lm = lane & 15;

  if (threadIdx.x < 16) acc[threadIdx.x] = 0.0f;

  // ---- stage 1: H1 = relu(X[m0:m0+16] @ W1^T + b1) -> LDS
  // wave w covers columns [32w, 32w+32): two 16-wide n-tiles, K = 64.
  ABFrag a0, a1;
  {
    const h8* ap0 = (const h8*)(X + (size_t)(m0 + lm) * KP1 + 0  + 8 * hi);
    a0.h[0] = ap0[0]; a0.h[1] = ap0[2];
    const h8* ap1 = (const h8*)(X + (size_t)(m0 + lm) * KP1 + 32 + 8 * hi);
    a1.h[0] = ap1[0]; a1.h[1] = ap1[2];
  }
  #pragma unroll
  for (int t = 0; t < 2; ++t) {
    const int n0 = w * 32 + t * 16;
    v8f c = {};
    ABFrag b;
    const h8* bp0 = (const h8*)(W1 + (size_t)(n0 + lm) * KP1 + 0 + 8 * hi);
    b.h[0] = bp0[0]; b.h[1] = bp0[2];
    c = wmma_f16(a0, b, c);
    const h8* bp1 = (const h8*)(W1 + (size_t)(n0 + lm) * KP1 + 32 + 8 * hi);
    b.h[0] = bp1[0]; b.h[1] = bp1[2];
    c = wmma_f16(a1, b, c);
    const int   n  = n0 + lm;
    const float bn = b1[n];
    #pragma unroll
    for (int r = 0; r < 8; ++r) {
      float v = c[r] + bn;                      // D VGPR r -> row r + 8*hi
      H1s[(r + 8 * hi) * LD1 + n] = (_Float16)(v > 0.0f ? v : 0.0f);
    }
  }
  __syncthreads();

  // ---- stage 2: H2 tile, wave w covers columns [16w, 16w+16), K = 256 from LDS
  v8f c2 = {};
  const int n0 = w * 16;
  #pragma unroll
  for (int ks = 0; ks < H1N; ks += 32) {
    ABFrag a, b;
    const h8* ap = (const h8*)(&H1s[lm * LD1 + ks + 8 * hi]);   // ds_load_b128
    a.h[0] = ap[0]; a.h[1] = ap[2];
    const h8* bp = (const h8*)(W2 + (size_t)(n0 + lm) * H1N + ks + 8 * hi);
    b.h[0] = bp[0]; b.h[1] = bp[2];
    c2 = wmma_f16(a, b, c2);
  }

  // ---- stage 3: out = sigmoid(relu(H2) . fw + fb + lin_out), fused reduction
  const int   n   = n0 + lm;
  const float bn  = b2[n];
  const float fwn = fw[n];
  float part[8];
  #pragma unroll
  for (int r = 0; r < 8; ++r) {
    float v = c2[r] + bn;
    v = (v > 0.0f ? v : 0.0f);
    part[r] = v * fwn;
  }
  // sum over this wave's 16 columns (xor of bits 0..3 stays within a half-wave)
  #pragma unroll
  for (int o = 1; o < 16; o <<= 1) {
    #pragma unroll
    for (int r = 0; r < 8; ++r) part[r] += __shfl_xor(part[r], o, 32);
  }
  if (lm == 0) {
    #pragma unroll
    for (int r = 0; r < 8; ++r)
      atomicAdd(&acc[r + 8 * hi], part[r]);     // ds_add_f32 across waves
  }
  __syncthreads();

  if (threadIdx.x < 16) {
    const int   m = m0 + threadIdx.x;
    const float z = acc[threadIdx.x] + fb[0] + lin_out[m];
    out[m] = 1.0f / (1.0f + __expf(-z));
  }
}

// ---------------- host side ----------------
extern "C" void kernel_launch(void* const* d_in, const int* in_sizes, int n_in,
                              void* d_out, int out_size, void* d_ws, size_t ws_size,
                              hipStream_t stream) {
  const float* ds   = (const float*)d_in[0];
  const int*   sp   = (const int*)  d_in[1];
  const float* emb  = (const float*)d_in[2];
  const float* lint = (const float*)d_in[3];
  const float* linw = (const float*)d_in[4];
  const float* linb = (const float*)d_in[5];
  const float* w1   = (const float*)d_in[6];
  const float* b1   = (const float*)d_in[7];
  const float* w2   = (const float*)d_in[8];
  const float* b2   = (const float*)d_in[9];
  const float* fw   = (const float*)d_in[10];
  const float* fb   = (const float*)d_in[11];
  float* out = (float*)d_out;

  char*  ws  = (char*)d_ws;
  size_t off = 0;
  auto alloc = [&](size_t bytes) -> char* {
    char* p = ws + off;
    off = (off + bytes + 255) & ~(size_t)255;
    return p;
  };
  _Float16* X    = (_Float16*)alloc((size_t)B_SZ * KP1 * sizeof(_Float16));
  _Float16* W1h  = (_Float16*)alloc((size_t)H1N * KP1 * sizeof(_Float16));
  _Float16* W2h  = (_Float16*)alloc((size_t)H2N * H1N * sizeof(_Float16));
  float*    LinO = (float*)   alloc((size_t)B_SZ * sizeof(float));
  (void)ws_size; (void)in_sizes; (void)n_in; (void)out_size;

  // weight conversions (tiny; stream-ordered)
  nfm_cvt_w1<<<(H1N * KP1) / 256, 256, 0, stream>>>(w1, W1h);
  nfm_cvt_w2<<<(H2N * H1N) / 256, 256, 0, stream>>>(w2, W2h);

  // gathers + bi-interaction: one wave per row
  nfm_features<<<B_SZ / 8, 256, 0, stream>>>(ds, sp, emb, lint, linw, linb, X, LinO);

  // fused MLP: one block per 16 rows
  nfm_mlp<<<B_SZ / 16, 256, 0, stream>>>(X, W1h, b1, W2h, b2, fw, fb, LinO, out);
}
